// LSTM2Encoder_40690520163127
// MI455X (gfx1250) — matrix operand order repaired and checked by reading, not compiled, not run on hardware
//
#include <hip/hip_runtime.h>

// ---------------------------------------------------------------------------
// Fused 2-layer zero-state LSTM + Linear for gfx1250, fp16 WMMA / fp32 accum.
//
// Roofline: 240 GFLOP useful vs ~270 MB compulsory HBM (11.6 us @ 23.3 TB/s)
// => needs ~20 PFLOP/s to be memory-bound. fp32 WMMA (K=4, 2048 FLOP/inst)
// can't get close; V_WMMA_F32_16X16X32_F16 (16384 FLOP/inst, fp32 accum) is
// the precision the chip rewards: ~8x fewer matrix instructions, half the
// LDS/L2 bytes. Weights are pre-converted to fp16 in d_ws each launch
// (i/g/o rows only; f gate and W_hh are dead since h_prev=c_prev=0).
//
// Main kernel: TM=128 tokens/block, 8 waves = 2 M-groups x 4 col-groups.
// Per col pass each wave holds acc[4 Mtiles][3 gates] fp32 accumulators;
// each B fragment feeds 4 WMMAs. LDS: 2 x 128x520 fp16 tiles = 266 KB.
// ---------------------------------------------------------------------------

typedef _Float16 v4h  __attribute__((ext_vector_type(4)));
typedef _Float16 v8h  __attribute__((ext_vector_type(8)));
typedef _Float16 v16h __attribute__((ext_vector_type(16)));
typedef float    v8f  __attribute__((ext_vector_type(8)));

#define HDIM    512
#define PITCH_H 520          // LDS row pitch (halves): 1040B -> 4-dword lane skew
#define TM      128          // tokens per block
#define MT      4            // 16-row M tiles per wave (2 wave groups x 4 = 8)
#define GSZ     (HDIM * HDIM)   // one gate block (halves) in converted ws

__device__ __forceinline__ v8f wmma_f16(v16h a, v16h b, v8f c) {
    // emits v_wmma_f32_16x16x32_f16 (probe-confirmed signature)
    return __builtin_amdgcn_wmma_f32_16x16x32_f16(
        false, a, false, b, (short)0, c, false, false);
}

__device__ __forceinline__ v16h cat8(v8h a, v8h b) {
    return __builtin_shufflevector(a, b, 0,1,2,3,4,5,6,7,8,9,10,11,12,13,14,15);
}

__device__ __forceinline__ float sig_f(float x) {
    return 1.0f / (1.0f + __expf(-x));
}
__device__ __forceinline__ float tanh_f(float x) {
    return 1.0f - 2.0f / (__expf(2.0f * x) + 1.0f);
}

// ---------------------------------------------------------------------------
// Prep: convert live weight rows to fp16 in ws.
//   ws halves layout: [L0: i,g,o][L1: i,g,o][lin] each gate block 512x512.
// ---------------------------------------------------------------------------
__global__ void cvt_weights_kernel(const float* __restrict__ W0,
                                   const float* __restrict__ W1,
                                   const float* __restrict__ WL,
                                   _Float16*    __restrict__ ws)
{
    const int idx = blockIdx.x * blockDim.x + threadIdx.x;   // 0 .. 7*GSZ-1
    if (idx >= 7 * GSZ) return;
    float v;
    if (idx < 6 * GSZ) {
        const int lidx = idx % (3 * GSZ);
        const float* src = (idx < 3 * GSZ) ? W0 : W1;
        const int g    = lidx / GSZ;          // 0=i,1=g,2=o
        const int rem  = lidx % GSZ;
        const int gate = (g == 0) ? 0 : (g == 1 ? 2 : 3);   // torch i,f,g,o rows
        v = src[(size_t)gate * GSZ + rem];
    } else {
        v = WL[idx - 6 * GSZ];
    }
    ws[idx] = (_Float16)v;
}

// ---------------------------------------------------------------------------
// One zero-state LSTM cell stage: dst = lstm(src @ W^T + b1 + b2)
// src/dst: LDS fp16 [TM][PITCH_H]; Wl: fp16 ws layer base ([3][512][512]).
// ---------------------------------------------------------------------------
__device__ __forceinline__ void lstm_stage(const _Float16* src, _Float16* dst,
                                           const _Float16* __restrict__ Wl,
                                           const float* __restrict__ b1,
                                           const float* __restrict__ b2,
                                           int lane, int wave)
{
    const int mrow  = lane & 15;
    const int hi    = lane >> 4;
    const int a_s   = hi * 8;       // A: upper lanes carry K+8 / K+24
    const int b_s   = hi * 8;       // B: upper lanes carry K+8 / K+24
    const int mbase = hi * 8;       // C/D: upper lanes hold rows 8..15
    const int mgrp  = wave >> 2;    // 2 M-groups
    const int cgrp  = wave & 3;     // 4 column groups

    const _Float16* abase = src + mrow * PITCH_H + a_s;

    for (int cc = 0; cc < 8; ++cc) {               // 8 x 64 = 512 columns
        const int ncol = cc * 64 + cgrp * 16 + mrow;
        const _Float16* bp_i = Wl + 0 * (size_t)GSZ + (size_t)ncol * HDIM + b_s;
        const _Float16* bp_g = Wl + 1 * (size_t)GSZ + (size_t)ncol * HDIM + b_s;
        const _Float16* bp_o = Wl + 2 * (size_t)GSZ + (size_t)ncol * HDIM + b_s;

        v8f acc[MT][3];
        #pragma unroll
        for (int mt = 0; mt < MT; ++mt)
            #pragma unroll
            for (int g = 0; g < 3; ++g)
                acc[mt][g] = (v8f)0.0f;

        #pragma unroll 2
        for (int k = 0; k < HDIM; k += 32) {
            v16h bi = cat8(*(const v8h*)(bp_i + k), *(const v8h*)(bp_i + k + 16));
            v16h bg = cat8(*(const v8h*)(bp_g + k), *(const v8h*)(bp_g + k + 16));
            v16h bo = cat8(*(const v8h*)(bp_o + k), *(const v8h*)(bp_o + k + 16));
            #pragma unroll
            for (int mt = 0; mt < MT; ++mt) {
                const _Float16* ap =
                    abase + (size_t)((mgrp * MT + mt) * 16) * PITCH_H + k;
                v16h a = cat8(*(const v8h*)ap, *(const v8h*)(ap + 16));
                acc[mt][0] = wmma_f16(a, bi, acc[mt][0]);
                acc[mt][1] = wmma_f16(a, bg, acc[mt][1]);
                acc[mt][2] = wmma_f16(a, bo, acc[mt][2]);
            }
        }

        const float bbi = b1[ncol]            + b2[ncol];
        const float bbg = b1[2 * HDIM + ncol] + b2[2 * HDIM + ncol];
        const float bbo = b1[3 * HDIM + ncol] + b2[3 * HDIM + ncol];
        #pragma unroll
        for (int mt = 0; mt < MT; ++mt) {
            #pragma unroll
            for (int j = 0; j < 8; ++j) {
                const int m = (mgrp * MT + mt) * 16 + mbase + j;
                float iv = sig_f (acc[mt][0][j] + bbi);
                float gv = tanh_f(acc[mt][1][j] + bbg);
                float ov = sig_f (acc[mt][2][j] + bbo);
                dst[m * PITCH_H + ncol] = (_Float16)(ov * tanh_f(iv * gv));
            }
        }
    }
}

__launch_bounds__(256)
__global__ void lstm2enc_kernel(const int*      __restrict__ ids,
                                const float*    __restrict__ emb,
                                const _Float16* __restrict__ ws,
                                const float*    __restrict__ b_ih0,
                                const float*    __restrict__ b_hh0,
                                const float*    __restrict__ b_ih1,
                                const float*    __restrict__ b_hh1,
                                const float*    __restrict__ b_lin,
                                float*          __restrict__ out,
                                int ntok)
{
    __shared__ _Float16 xbuf[TM * PITCH_H];   // x, then reused for h1
    __shared__ _Float16 hbuf[TM * PITCH_H];   // h0

    const int tid  = threadIdx.x;
    const int lane = tid & 31;
    const int wave = tid >> 5;
    const int tok0 = blockIdx.x * TM;

    // ---- gather TM embedding rows, convert fp32 -> fp16 into LDS ----
    {
        const int c = tid & 15;                          // float4 column group
        for (int r = tid >> 4; r < TM; r += 16) {
            int tok = tok0 + r;
            if (tok >= ntok) tok = ntok - 1;
            const int id = ids[tok];
            const float4* s = (const float4*)(emb + (size_t)id * HDIM) + c;
            _Float16* d = xbuf + r * PITCH_H;
            #pragma unroll
            for (int i = 0; i < 8; ++i) {
                float4 v = s[i * 16];
                v4h h = { (_Float16)v.x, (_Float16)v.y,
                          (_Float16)v.z, (_Float16)v.w };
                *(v4h*)(d + (c + i * 16) * 4) = h;       // 8B-aligned ds store
            }
        }
    }
    __syncthreads();

    lstm_stage(xbuf, hbuf, ws,           b_ih0, b_hh0, lane, wave);  // x  -> h0
    __syncthreads();
    lstm_stage(hbuf, xbuf, ws + 3 * (size_t)GSZ, b_ih1, b_hh1, lane, wave); // h0 -> h1
    __syncthreads();

    // ---- final linear + relu ----
    {
        const int mrow  = lane & 15;
        const int hi    = lane >> 4;
        const int a_s   = hi * 8;
        const int b_s   = hi * 8;
        const int mbase = hi * 8;
        const int mgrp  = wave >> 2;
        const int cgrp  = wave & 3;
        const _Float16* abase = xbuf + mrow * PITCH_H + a_s;
        const _Float16* WL = ws + 6 * (size_t)GSZ;

        for (int cc = 0; cc < 8; ++cc) {
            const int ncol = cc * 64 + cgrp * 16 + mrow;
            const _Float16* bp = WL + (size_t)ncol * HDIM + b_s;

            v8f acc[MT];
            #pragma unroll
            for (int mt = 0; mt < MT; ++mt) acc[mt] = (v8f)0.0f;

            #pragma unroll 2
            for (int k = 0; k < HDIM; k += 32) {
                v16h b = cat8(*(const v8h*)(bp + k), *(const v8h*)(bp + k + 16));
                #pragma unroll
                for (int mt = 0; mt < MT; ++mt) {
                    const _Float16* ap =
                        abase + (size_t)((mgrp * MT + mt) * 16) * PITCH_H + k;
                    v16h a = cat8(*(const v8h*)ap, *(const v8h*)(ap + 16));
                    acc[mt] = wmma_f16(a, b, acc[mt]);
                }
            }

            const float bb = b_lin[ncol];
            #pragma unroll
            for (int mt = 0; mt < MT; ++mt) {
                #pragma unroll
                for (int j = 0; j < 8; ++j) {
                    const int m   = (mgrp * MT + mt) * 16 + mbase + j;
                    const int tok = tok0 + m;
                    float v = acc[mt][j] + bb;
                    v = v > 0.0f ? v : 0.0f;
                    if (tok < ntok) out[(size_t)tok * HDIM + ncol] = v;
                }
            }
        }
    }
}

extern "C" void kernel_launch(void* const* d_in, const int* in_sizes, int n_in,
                              void* d_out, int out_size, void* d_ws, size_t ws_size,
                              hipStream_t stream)
{
    // 0 ids, 1 emb, 2 W_ih0, 3 W_hh0(dead), 4 b_ih0, 5 b_hh0,
    // 6 W_ih1, 7 W_hh1(dead), 8 b_ih1, 9 b_hh1, 10 W_lin, 11 b_lin
    const int*   ids   = (const int*)  d_in[0];
    const float* emb   = (const float*)d_in[1];
    const float* W_ih0 = (const float*)d_in[2];
    const float* b_ih0 = (const float*)d_in[4];
    const float* b_hh0 = (const float*)d_in[5];
    const float* W_ih1 = (const float*)d_in[6];
    const float* b_ih1 = (const float*)d_in[8];
    const float* b_hh1 = (const float*)d_in[9];
    const float* W_lin = (const float*)d_in[10];
    const float* b_lin = (const float*)d_in[11];
    float*    out  = (float*)d_out;
    _Float16* ws16 = (_Float16*)d_ws;

    const int ntok = in_sizes[0];

    // 1) fp32 -> fp16 weight repack (7 * 512*512 halves = 3.67 MB in d_ws)
    {
        const int total  = 7 * GSZ;
        const int blocks = (total + 255) / 256;
        cvt_weights_kernel<<<dim3(blocks), dim3(256), 0, stream>>>(
            W_ih0, W_ih1, W_lin, ws16);
    }

    // 2) fused LSTM x2 + linear
    {
        const int blocks = (ntok + TM - 1) / TM;
        lstm2enc_kernel<<<dim3(blocks), dim3(256), 0, stream>>>(
            ids, emb, ws16, b_ih0, b_hh0, b_ih1, b_hh1, b_lin, out, ntok);
    }
}
